// GatedDeltaNetChunkwise_37417755083581
// MI455X (gfx1250) — compile-verified, matrix operand order checked by reading, not run on hardware
//
#include <hip/hip_runtime.h>
#include <hip/hip_bf16.h>

// ---------------------------------------------------------------------------
// Gated DeltaNet forward for MI455X (gfx1250, wave32, WMMA + async-to-LDS)
//   dims: B=1, T=512, H=2048, HK=16, DK=128, HV=32, DV=128, VE=256, KC=4
// ---------------------------------------------------------------------------

#define Tt  512
#define Hh  2048
#define NQK 2048      // HK*DK
#define NVV 4096      // HV*DV
#define HKh 16
#define DKd 128
#define VEe 256

// gfx1250 async LDS-copy path (inline asm; flip to 0 if assembler rejects)
#define USE_ASYNC_ASM 1

typedef __attribute__((ext_vector_type(16))) __bf16         v16bf;
typedef __attribute__((ext_vector_type(8)))  __bf16         v8bf;
typedef __attribute__((ext_vector_type(8)))  unsigned short v8us;
typedef __attribute__((ext_vector_type(8)))  float          v8f;

// ---- fp32 -> bf16 (round to nearest even) ---------------------------------
__device__ __forceinline__ unsigned short f2bf(float f) {
    union { float f; unsigned u; } cv; cv.f = f;
    unsigned u = cv.u;
    unsigned r = (u + 0x7FFFu + ((u >> 16) & 1u)) >> 16;
    return (unsigned short)r;
}

__global__ void f32_to_bf16_kernel(const float* __restrict__ in,
                                   unsigned short* __restrict__ out, size_t n) {
    size_t i = (size_t)blockIdx.x * blockDim.x + threadIdx.x;
    if (i < n) out[i] = f2bf(in[i]);
}

// ---- pack a KxN fp32 weight matrix into exact WMMA B-fragment order -------
// frag element j of lane L for k-block kb, n-tile nt maps to
//   k = kb*32 + (j>>3)*16 + (L>>4)*8 + (j&7),  n = nt*16 + (L&15)
// stored contiguously: F[(((nt*nKb)+kb)*32 + L)*16 + j]
__global__ void pack_wfrag_kernel(const float* __restrict__ W,
                                  unsigned short* __restrict__ F,
                                  int K, int N) {
    size_t idx = (size_t)blockIdx.x * blockDim.x + threadIdx.x;
    if (idx >= (size_t)K * N) return;
    int j    = (int)(idx & 15);
    int lane = (int)((idx >> 4) & 31);
    size_t rest = idx >> 9;               // nt*nKb + kb
    int nKb = K >> 5;
    int kb  = (int)(rest % nKb);
    int nt  = (int)(rest / nKb);
    int hi = lane >> 4, r = lane & 15;
    int k = kb * 32 + ((j >> 3) << 4) + hi * 8 + (j & 7);
    int n = nt * 16 + r;
    F[idx] = f2bf(W[(size_t)k * N + n]);
}

// ---- bf16 WMMA GEMM, 16x64 per wave, LDS-staged A, async double buffer ----
// Block = 8 waves sharing one 16-row m-tile; wave w owns n-tiles [w*4, w*4+4).
// A tile (16x32 bf16 = 1KB) staged in LDS by threads 0..63 per k-step,
// double-buffered with global_load_async_to_lds_b128 + s_wait_asynccnt.
__global__ void __launch_bounds__(256)
gemm_bf16_wmma_kernel(const unsigned short* __restrict__ Abf,
                      const unsigned short* __restrict__ Bfrag,
                      float* __restrict__ C, int M, int N, int K) {
    __shared__ __align__(16) unsigned short Atile[2][16 * 32];

    int tid  = threadIdx.x;
    int lane = tid & 31;
    int wave = tid >> 5;
    int r  = lane & 15;
    int hi = lane >> 4;

    int nBlk = N >> 9;                    // N / 512
    int mb = blockIdx.x / nBlk;
    int nb = blockIdx.x % nBlk;
    int nKb = K >> 5;
    int nt0 = nb * 32 + wave * 4;         // first of this wave's 4 n-tiles

    // cooperative A-tile copy: threads 0..63, 16B (8 bf16) each
    int crow = tid >> 2, cchk = tid & 3;  // row 0..15 (valid for tid<64), chunk 0..3
    const unsigned short* asrc =
        Abf + (size_t)(mb * 16 + (crow & 15)) * K + cchk * 8;
    unsigned ldsoff[2];
    ldsoff[0] = (unsigned)(uintptr_t)&Atile[0][(crow & 15) * 32 + cchk * 8];
    ldsoff[1] = (unsigned)(uintptr_t)&Atile[1][(crow & 15) * 32 + cchk * 8];

    // B fragment stream pointers (one per owned n-tile)
    const unsigned short* bp[4];
#pragma unroll
    for (int tt = 0; tt < 4; ++tt)
        bp[tt] = Bfrag + (((size_t)(nt0 + tt) * nKb) * 32 + lane) * 16;

    v8f a0 = {0.f,0.f,0.f,0.f,0.f,0.f,0.f,0.f};
    v8f a1 = a0, a2 = a0, a3 = a0;

    // ---- prologue: stage kb = 0 into buffer 0 ----
    if (tid < 64) {
        const unsigned short* g = asrc;
#if USE_ASYNC_ASM
        asm volatile("global_load_async_to_lds_b128 %0, %1, off"
                     :: "v"(ldsoff[0]), "v"(g) : "memory");
#else
        *(v8us*)(void*)&Atile[0][(crow & 15) * 32 + cchk * 8] =
            *(const v8us*)(const void*)g;
#endif
    }
#if USE_ASYNC_ASM
    asm volatile("s_wait_asynccnt 0x0" ::: "memory");
#endif
    __syncthreads();

    int cur = 0;
    for (int kb = 0; kb < nKb; ++kb) {
        // stage kb+1 into the other buffer (it was last read in kb-1)
        if (kb + 1 < nKb) {
            if (tid < 64) {
                const unsigned short* g = asrc + (size_t)(kb + 1) * 32;
#if USE_ASYNC_ASM
                asm volatile("global_load_async_to_lds_b128 %0, %1, off"
                             :: "v"(ldsoff[cur ^ 1]), "v"(g) : "memory");
#else
                *(v8us*)(void*)&Atile[cur ^ 1][(crow & 15) * 32 + cchk * 8] =
                    *(const v8us*)(const void*)g;
#endif
            }
        }

        // A fragment from LDS (two ds_load_b128 per lane)
        const unsigned short* ap = &Atile[cur][r * 32 + hi * 8];
        union { v8us u; v8bf b; } lo, hic;
        lo.u  = *(const v8us*)(const void*)ap;
        hic.u = *(const v8us*)(const void*)(ap + 16);
        v16bf a = {};
#pragma unroll
        for (int i = 0; i < 8; ++i) { a[i] = lo.b[i]; a[i + 8] = hic.b[i]; }

        // 4 streamed B fragments -> 4 WMMAs sharing the same A fragment
        v16bf b0 = *(const v16bf*)(const void*)(bp[0] + (size_t)kb * 512);
        v16bf b1 = *(const v16bf*)(const void*)(bp[1] + (size_t)kb * 512);
        v16bf b2 = *(const v16bf*)(const void*)(bp[2] + (size_t)kb * 512);
        v16bf b3 = *(const v16bf*)(const void*)(bp[3] + (size_t)kb * 512);
        if (kb + 1 < nKb) {
            __builtin_prefetch(bp[0] + (size_t)(kb + 1) * 512, 0, 0);
            __builtin_prefetch(bp[1] + (size_t)(kb + 1) * 512, 0, 0);
            __builtin_prefetch(bp[2] + (size_t)(kb + 1) * 512, 0, 0);
            __builtin_prefetch(bp[3] + (size_t)(kb + 1) * 512, 0, 0);
        }
        a0 = __builtin_amdgcn_wmma_f32_16x16x32_bf16(false, a, false, b0, (short)0, a0, false, false);
        a1 = __builtin_amdgcn_wmma_f32_16x16x32_bf16(false, a, false, b1, (short)0, a1, false, false);
        a2 = __builtin_amdgcn_wmma_f32_16x16x32_bf16(false, a, false, b2, (short)0, a2, false, false);
        a3 = __builtin_amdgcn_wmma_f32_16x16x32_bf16(false, a, false, b3, (short)0, a3, false, false);

#if USE_ASYNC_ASM
        asm volatile("s_wait_asynccnt 0x0" ::: "memory");
#endif
        __syncthreads();   // next buffer filled AND this buffer fully read
        cur ^= 1;
    }

    // C/D layout: VGPR i -> M = mb*16 + i + hi*8, N = nt*16 + r
#pragma unroll
    for (int i = 0; i < 8; ++i) {
        size_t rowoff = (size_t)(mb * 16 + i + hi * 8) * N + r;
        C[rowoff + (nt0 + 0) * 16] = a0[i];
        C[rowoff + (nt0 + 1) * 16] = a1[i];
        C[rowoff + (nt0 + 2) * 16] = a2[i];
        C[rowoff + (nt0 + 3) * 16] = a3[i];
    }
}

// ---- alpha/beta: sigmoid(x @ Wa/Wb), one block per timestep ---------------
__global__ void __launch_bounds__(256)
ab_proj_sigmoid_kernel(const float* __restrict__ X,
                       const float* __restrict__ Wa,
                       const float* __restrict__ Wb,
                       float* __restrict__ AB) {
    __shared__ float red[256];
    int t = blockIdx.x, tid = threadIdx.x;
    int outIdx = tid >> 3;      // 0..31 : 16 alpha + 16 beta
    int part = tid & 7;
    const float* W = (outIdx < 16) ? Wa : Wb;
    int hcol = outIdx & 15;
    const float* xr = X + (size_t)t * Hh;
    float s = 0.f;
    for (int k = part * 256; k < part * 256 + 256; ++k)
        s += xr[k] * W[k * HKh + hcol];
    red[tid] = s;
    __syncthreads();
    if (part == 0) {
        float tot = 0.f;
#pragma unroll
        for (int i = 0; i < 8; ++i) tot += red[(outIdx << 3) + i];
        float sg = 1.f / (1.f + __expf(-tot));
        if (outIdx < 16) AB[t * HKh + hcol] = sg;
        else             AB[Tt * HKh + t * HKh + hcol] = sg;
    }
}

// ---- causal conv(K=4) + SiLU + per-head L2 norm (q, k): one wave/(t,head) -
__global__ void __launch_bounds__(256)
conv_silu_norm_kernel(const float* __restrict__ X, const float* __restrict__ w,
                      const float* __restrict__ bias, float* __restrict__ Y,
                      int T, int C) {
    int gwave = (blockIdx.x * blockDim.x + threadIdx.x) >> 5;
    int lane  = threadIdx.x & 31;
    int heads = C >> 7;
    if (gwave >= T * heads) return;
    int h = gwave % heads, t = gwave / heads;
    float y[4], ss = 0.f;
#pragma unroll
    for (int i = 0; i < 4; ++i) {
        int c = h * 128 + lane * 4 + i;
        float acc = bias[c];
#pragma unroll
        for (int j = 0; j < 4; ++j) {
            int tt = t + j - 3;
            if (tt >= 0) acc += w[c * 4 + j] * X[(size_t)tt * C + c];
        }
        acc = acc / (1.f + __expf(-acc));          // SiLU
        y[i] = acc; ss += acc * acc;
    }
#pragma unroll
    for (int off = 16; off; off >>= 1) ss += __shfl_xor(ss, off, 32);
    float scale = 1.f / fmaxf(sqrtf(ss), 1e-12f);
#pragma unroll
    for (int i = 0; i < 4; ++i)
        Y[(size_t)t * C + h * 128 + lane * 4 + i] = y[i] * scale;
}

// ---- causal conv(K=4) + SiLU elementwise (v) ------------------------------
__global__ void conv_silu_kernel(const float* __restrict__ X,
                                 const float* __restrict__ w,
                                 const float* __restrict__ bias,
                                 float* __restrict__ Y, int T, int C) {
    size_t idx = (size_t)blockIdx.x * blockDim.x + threadIdx.x;
    if (idx >= (size_t)T * C) return;
    int c = (int)(idx % C), t = (int)(idx / C);
    float acc = bias[c];
#pragma unroll
    for (int j = 0; j < 4; ++j) {
        int tt = t + j - 3;
        if (tt >= 0) acc += w[c * 4 + j] * X[(size_t)tt * C + c];
    }
    Y[idx] = acc / (1.f + __expf(-acc));
}

// ---- gated delta-rule scan: one workgroup per head, state lives in LDS ----
// state stored transposed S[v][k] with row stride 129 (bank-conflict free).
#define SROW 129
__global__ void __launch_bounds__(256)
recurrence_kernel(const float* __restrict__ Q, const float* __restrict__ Kk,
                  const float* __restrict__ V, const float* __restrict__ AB,
                  float* __restrict__ O, float* __restrict__ stateOut) {
    extern __shared__ float sm[];
    float* S  = sm;                       // 256 * 129
    float* sk = sm + 256 * SROW;          // 128
    float* sq = sk + 128;                 // 128
    float* sv = sq + 128;                 // 256
    int h = blockIdx.x, tid = threadIdx.x;

    for (int i = tid; i < 256 * SROW; i += 256) S[i] = 0.f;
    __syncthreads();

    float* Srow = S + tid * SROW;
    for (int t = 0; t < Tt; ++t) {
        if (tid < 128) {
            sk[tid] = Kk[(size_t)t * NQK + h * DKd + tid];
            sq[tid] = Q [(size_t)t * NQK + h * DKd + tid];
        }
        sv[tid] = V[(size_t)t * NVV + h * VEe + tid];
        __syncthreads();
        float alpha = AB[t * HKh + h];
        float beta  = AB[Tt * HKh + t * HKh + h];
        float dot = 0.f;
#pragma unroll 8
        for (int k = 0; k < DKd; ++k) dot += Srow[k] * sk[k];
        float err = sv[tid] - alpha * dot;       // v - k . (alpha*S)
        float be  = beta * err;
        float o = 0.f;
#pragma unroll 8
        for (int k = 0; k < DKd; ++k) {
            float s = alpha * Srow[k] + be * sk[k];
            Srow[k] = s;
            o += sq[k] * s;
        }
        O[(size_t)t * NVV + h * VEe + tid] = o;
        __syncthreads();
    }
    // final state: layout [HK][DK][VE]
    for (int i = tid; i < DKd * VEe; i += 256) {
        int k = i >> 8, v = i & 255;
        stateOut[(size_t)h * DKd * VEe + i] = S[v * SROW + k];
    }
}

// ---------------------------------------------------------------------------
extern "C" void kernel_launch(void* const* d_in, const int* in_sizes, int n_in,
                              void* d_out, int out_size, void* d_ws, size_t ws_size,
                              hipStream_t stream) {
    (void)in_sizes; (void)n_in; (void)out_size; (void)ws_size;

    const float* x   = (const float*)d_in[0];
    const float* Wq  = (const float*)d_in[1];
    const float* Wk  = (const float*)d_in[2];
    const float* Wv  = (const float*)d_in[3];
    const float* Wo  = (const float*)d_in[4];
    const float* Wa  = (const float*)d_in[5];
    const float* Wb  = (const float*)d_in[6];
    const float* qcw = (const float*)d_in[7];
    const float* qcb = (const float*)d_in[8];
    const float* kcw = (const float*)d_in[9];
    const float* kcb = (const float*)d_in[10];
    const float* vcw = (const float*)d_in[11];
    const float* vcb = (const float*)d_in[12];

    char* wsp = (char*)d_ws;
    auto carve = [&](size_t bytes) -> void* {
        void* p = (void*)wsp;
        wsp += (bytes + 255) & ~(size_t)255;
        return p;
    };

    unsigned short* xbf  = (unsigned short*)carve((size_t)Tt * Hh * 2);
    unsigned short* WqF  = (unsigned short*)carve((size_t)Hh * NQK * 2);
    unsigned short* WkF  = (unsigned short*)carve((size_t)Hh * NQK * 2);
    unsigned short* WvF  = (unsigned short*)carve((size_t)Hh * NVV * 2);
    unsigned short* WoF  = (unsigned short*)carve((size_t)NVV * Hh * 2);
    float* qproj = (float*)carve((size_t)Tt * NQK * 4);
    float* kproj = (float*)carve((size_t)Tt * NQK * 4);
    float* vproj = (float*)carve((size_t)Tt * NVV * 4);
    float* q2    = (float*)carve((size_t)Tt * NQK * 4);
    float* k2    = (float*)carve((size_t)Tt * NQK * 4);
    float* v2    = (float*)carve((size_t)Tt * NVV * 4);
    float* AB    = (float*)carve((size_t)2 * Tt * HKh * 4);
    float* obuf  = (float*)carve((size_t)Tt * NVV * 4);
    unsigned short* obf = (unsigned short*)carve((size_t)Tt * NVV * 2);

    float* outMain  = (float*)d_out;                       // [T, H]
    float* outState = (float*)d_out + (size_t)Tt * Hh;     // [HK, DK, VE]

    // 1) conversions / weight fragment packing
    {
        size_t n = (size_t)Tt * Hh;
        f32_to_bf16_kernel<<<(unsigned)((n + 255) / 256), 256, 0, stream>>>(x, xbf, n);
    }
    pack_wfrag_kernel<<<(unsigned)(((size_t)Hh * NQK + 255) / 256), 256, 0, stream>>>(Wq, WqF, Hh, NQK);
    pack_wfrag_kernel<<<(unsigned)(((size_t)Hh * NQK + 255) / 256), 256, 0, stream>>>(Wk, WkF, Hh, NQK);
    pack_wfrag_kernel<<<(unsigned)(((size_t)Hh * NVV + 255) / 256), 256, 0, stream>>>(Wv, WvF, Hh, NVV);
    pack_wfrag_kernel<<<(unsigned)(((size_t)NVV * Hh + 255) / 256), 256, 0, stream>>>(Wo, WoF, NVV, Hh);

    // 2) projection GEMMs (bf16 WMMA, fp32 accumulate); grid = (M/16)*(N/512)
    gemm_bf16_wmma_kernel<<<(Tt / 16) * (NQK / 512), 256, 0, stream>>>(xbf, WqF, qproj, Tt, NQK, Hh);
    gemm_bf16_wmma_kernel<<<(Tt / 16) * (NQK / 512), 256, 0, stream>>>(xbf, WkF, kproj, Tt, NQK, Hh);
    gemm_bf16_wmma_kernel<<<(Tt / 16) * (NVV / 512), 256, 0, stream>>>(xbf, WvF, vproj, Tt, NVV, Hh);

    // 3) gates
    ab_proj_sigmoid_kernel<<<Tt, 256, 0, stream>>>(x, Wa, Wb, AB);

    // 4) conv + SiLU (+ L2 norm for q,k)
    conv_silu_norm_kernel<<<(Tt * HKh * 32) / 256, 256, 0, stream>>>(qproj, qcw, qcb, q2, Tt, NQK);
    conv_silu_norm_kernel<<<(Tt * HKh * 32) / 256, 256, 0, stream>>>(kproj, kcw, kcb, k2, Tt, NQK);
    conv_silu_kernel<<<(unsigned)(((size_t)Tt * NVV + 255) / 256), 256, 0, stream>>>(vproj, vcw, vcb, v2, Tt, NVV);

    // 5) gated delta-rule scan: 16 workgroups, 129KB state in LDS per head
    size_t smem = (size_t)(256 * SROW + 128 + 128 + 256) * sizeof(float);
    recurrence_kernel<<<HKh, 256, smem, stream>>>(q2, k2, v2, AB, obuf, outState);

    // 6) output projection
    {
        size_t n = (size_t)Tt * NVV;
        f32_to_bf16_kernel<<<(unsigned)((n + 255) / 256), 256, 0, stream>>>(obuf, obf, n);
    }
    gemm_bf16_wmma_kernel<<<(Tt / 16) * (Hh / 512), 256, 0, stream>>>(obf, WoF, outMain, Tt, Hh, NVV);
}
